// Qwen3MoeDecoderLayer_36283883716955
// MI455X (gfx1250) — matrix operand order repaired and checked by reading, not compiled, not run on hardware
//
#include <hip/hip_runtime.h>
#include <cstdint>
#include <cstddef>

// ---------------- problem dims ----------------
constexpr int   kT    = 1024;
constexpr int   kH    = 2048;
constexpr int   kNH   = 16;
constexpr int   kKVH  = 4;
constexpr int   kD    = 128;
constexpr int   kE    = 32;
constexpr int   kTopK = 4;
constexpr int   kI    = 768;
constexpr int   kQKVW = (kNH + 2 * kKVH) * kD;   // 3072
constexpr int   kSlots = kT * kTopK;             // 4096 (always exactly this)
constexpr float kEPS  = 1e-6f;
constexpr float kLogTheta = 9.210340371976184f;  // ln(10000)

// ---------------- WMMA plumbing ----------------
typedef __attribute__((ext_vector_type(16))) __bf16 v16bf;
typedef __attribute__((ext_vector_type(8)))  float  v8f;

union Frag { unsigned int u[8]; v16bf v; };

// NOTE: by-value args — passing by reference takes the aggregate's address and
// can block SROA, which was spilling staged tiles to scratch.
__device__ __forceinline__ v8f wmma_bf16(Frag a, Frag b, v8f c) {
  // v_wmma_f32_16x16x32_bf16: D = A(16x32) * B(32x16) + C(16x16 f32)
  return __builtin_amdgcn_wmma_f32_16x16x32_bf16(
      false, a.v, false, b.v, (short)0, c, false, false);
}

__device__ __forceinline__ unsigned short f2bf(float f) {  // RNE float -> bf16 bits
  unsigned int u = __float_as_uint(f);
  u += 0x7FFFu + ((u >> 16) & 1u);
  return (unsigned short)(u >> 16);
}
__device__ __forceinline__ float bf2f(unsigned short b) {
  return __uint_as_float(((unsigned int)b) << 16);
}
__device__ __forceinline__ unsigned int pack2bf(float a, float b) {
  return (unsigned int)f2bf(a) | ((unsigned int)f2bf(b) << 16);
}

// A-fragment (16x32 bf16, MxK): lane m=l%16, hi=l/16.
// dword j holds K pair {kd,kd+1}, kd = 2j + (j>=4 ? 8:0) + hi*8
__device__ __forceinline__ int afrag_k(int j, int hi) { return 2 * j + ((j >= 4) ? 8 : 0) + hi * 8; }
// B-fragment (32x16 bf16, KxN): lane n=l%16; dword j -> kd = 2j + (j>=4 ? 8:0) + hi*8
__device__ __forceinline__ int bfrag_k(int j, int hi) { return 2 * j + ((j >= 4) ? 8 : 0) + hi * 8; }

// ---------------- GEMM common: C[M,N](f32) = A_bf16[M,K] * W_f32[N,K]^T -------
// 128x64 block tile, BK=64, double-buffered LDS with register staging.
constexpr int BM = 128, BN = 64, BK = 64, LDSS = 72;  // LDS row stride (elems)

// 8 WMMAs from one staged tile (2 k-steps of 32).
__device__ __forceinline__ void compute_tile(
    const unsigned short* AsBuf, const unsigned short* BsBuf,
    int wave, int l16, int hi,
    v8f& acc0, v8f& acc1, v8f& acc2, v8f& acc3) {
#pragma unroll
  for (int ks = 0; ks < BK; ks += 32) {
    Frag a, b0, b1, b2, b3;
#pragma unroll
    for (int j = 0; j < 8; j++) {
      a.u[j]  = *(const unsigned int*)&AsBuf[(wave * 16 + l16) * LDSS + ks + afrag_k(j, hi)];
      b0.u[j] = *(const unsigned int*)&BsBuf[(l16)      * LDSS + ks + bfrag_k(j, hi)];
      b1.u[j] = *(const unsigned int*)&BsBuf[(16 + l16) * LDSS + ks + bfrag_k(j, hi)];
      b2.u[j] = *(const unsigned int*)&BsBuf[(32 + l16) * LDSS + ks + bfrag_k(j, hi)];
      b3.u[j] = *(const unsigned int*)&BsBuf[(48 + l16) * LDSS + ks + bfrag_k(j, hi)];
    }
    acc0 = wmma_bf16(a, b0, acc0);
    acc1 = wmma_bf16(a, b1, acc1);
    acc2 = wmma_bf16(a, b2, acc2);
    acc3 = wmma_bf16(a, b3, acc3);
  }
}

// Named-scalar tile staging (no arrays / lambdas -> guaranteed SSA promotion).
#define LOAD_TILE(k0)                              \
  do {                                             \
    ar0 = *(const uint4*)(aP0 + (k0));             \
    ar1 = *(const uint4*)(aP1 + (k0));             \
    ar2 = *(const uint4*)(aP2 + (k0));             \
    ar3 = *(const uint4*)(aP3 + (k0));             \
    wr0 = *(const float4*)(wP0 + (k0));            \
    wr1 = *(const float4*)(wP1 + (k0));            \
    wr2 = *(const float4*)(wP2 + (k0));            \
    wr3 = *(const float4*)(wP3 + (k0));            \
  } while (0)

#define STAGE_TILE(bb)                                                        \
  do {                                                                        \
    *(uint4*)&As[bb][aL]             = ar0;                                   \
    *(uint4*)&As[bb][aL + 32 * LDSS] = ar1;                                   \
    *(uint4*)&As[bb][aL + 64 * LDSS] = ar2;                                   \
    *(uint4*)&As[bb][aL + 96 * LDSS] = ar3;                                   \
    *(unsigned int*)&Bs[bb][wL]                 = pack2bf(wr0.x, wr0.y);      \
    *(unsigned int*)&Bs[bb][wL + 2]             = pack2bf(wr0.z, wr0.w);      \
    *(unsigned int*)&Bs[bb][wL + 16 * LDSS]     = pack2bf(wr1.x, wr1.y);      \
    *(unsigned int*)&Bs[bb][wL + 16 * LDSS + 2] = pack2bf(wr1.z, wr1.w);      \
    *(unsigned int*)&Bs[bb][wL + 32 * LDSS]     = pack2bf(wr2.x, wr2.y);      \
    *(unsigned int*)&Bs[bb][wL + 32 * LDSS + 2] = pack2bf(wr2.z, wr2.w);      \
    *(unsigned int*)&Bs[bb][wL + 48 * LDSS]     = pack2bf(wr3.x, wr3.y);      \
    *(unsigned int*)&Bs[bb][wL + 48 * LDSS + 2] = pack2bf(wr3.z, wr3.w);      \
  } while (0)

__global__ __launch_bounds__(256)
__attribute__((amdgpu_waves_per_eu(2)))
void gemm_bf16_kernel(
    const unsigned short* __restrict__ A, const float* __restrict__ W,
    float* __restrict__ C, int M, int N, int K) {
  __shared__ unsigned short As[2][BM * LDSS];
  __shared__ unsigned short Bs[2][BN * LDSS];
  const int tid = threadIdx.x, wave = tid >> 5, lane = tid & 31;
  const int hi = lane >> 4, l16 = lane & 15;
  const int m0 = blockIdx.y * BM, n0 = blockIdx.x * BN;
  const int aRow = tid >> 3, aCol = (tid & 7) * 8;    // A: 32-row slices
  const int wRow = tid >> 4, wCol = (tid & 15) * 4;   // W: 16-row slices
  const unsigned short* aP0 = A + (size_t)(m0 + aRow) * K + aCol;
  const unsigned short* aP1 = aP0 + (size_t)32 * K;
  const unsigned short* aP2 = aP0 + (size_t)64 * K;
  const unsigned short* aP3 = aP0 + (size_t)96 * K;
  const float* wP0 = W + (size_t)(n0 + wRow) * K + wCol;
  const float* wP1 = wP0 + (size_t)16 * K;
  const float* wP2 = wP0 + (size_t)32 * K;
  const float* wP3 = wP0 + (size_t)48 * K;
  const int aL = aRow * LDSS + aCol;
  const int wL = wRow * LDSS + wCol;

  v8f acc0 = {}, acc1 = {}, acc2 = {}, acc3 = {};
  uint4  ar0, ar1, ar2, ar3;
  float4 wr0, wr1, wr2, wr3;

  LOAD_TILE(0);
  int buf = 0;
  for (int k0 = 0; k0 < K; k0 += BK) {
    __syncthreads();              // prior readers of lds[buf] are done
    STAGE_TILE(buf);
    if (k0 + BK < K) {            // issue next tile: overlaps with compute below
      LOAD_TILE(k0 + BK);
      if (k0 + 2 * BK < K)        // stream weights ahead (global_prefetch_b8)
        __builtin_prefetch(wP0 + (size_t)k0 + 2 * BK, 0, 1);
    }
    __syncthreads();
    compute_tile(&As[buf][0], &Bs[buf][0], wave, l16, hi, acc0, acc1, acc2, acc3);
    buf ^= 1;
  }
#pragma unroll
  for (int t = 0; t < 4; t++) {
    v8f av = (t == 0) ? acc0 : (t == 1) ? acc1 : (t == 2) ? acc2 : acc3;
#pragma unroll
    for (int r = 0; r < 8; r++) {
      int m = m0 + wave * 16 + r + hi * 8;
      int n = n0 + t * 16 + l16;
      C[(size_t)m * N + n] = av[r];
    }
  }
}

// ---------------- grouped MoE GEMM over gathered token slots ----------------
template <bool GATHER, bool WEIGHTED>
__global__ __launch_bounds__(256)
__attribute__((amdgpu_waves_per_eu(2)))
void moe_gemm_kernel(
    const unsigned short* __restrict__ A,  // GATHER: [T,K] gathered via perm; else: [slot,K]
    const float* __restrict__ Wbase,       // [E][N,K]
    float* __restrict__ C,                 // [slot, N]
    int N, int K, size_t wstride,
    const int* __restrict__ counts, const int* __restrict__ offsets,
    const int* __restrict__ perm, const float* __restrict__ slot_wt) {
  const int e = blockIdx.z;
  const int cnt = counts[e];
  const int m0 = blockIdx.y * BM;
  if (m0 >= cnt) return;
  const int off = offsets[e];
  const float* W = Wbase + (size_t)e * wstride;
  __shared__ unsigned short As[2][BM * LDSS];
  __shared__ unsigned short Bs[2][BN * LDSS];
  const int tid = threadIdx.x, wave = tid >> 5, lane = tid & 31;
  const int hi = lane >> 4, l16 = lane & 15;
  const int n0 = blockIdx.x * BN;
  const int aRow = tid >> 3, aCol = (tid & 7) * 8;
  const int wRow = tid >> 4, wCol = (tid & 15) * 4;
  // resolve this thread's 4 gathered source rows once
  int rr0 = m0 + aRow, rr1 = rr0 + 32, rr2 = rr0 + 64, rr3 = rr0 + 96;
  int g0, g1, g2, g3;
  if (GATHER) {
    g0 = (rr0 < cnt) ? perm[off + rr0] : perm[off];
    g1 = (rr1 < cnt) ? perm[off + rr1] : perm[off];
    g2 = (rr2 < cnt) ? perm[off + rr2] : perm[off];
    g3 = (rr3 < cnt) ? perm[off + rr3] : perm[off];
  } else {
    g0 = (rr0 < cnt) ? (off + rr0) : off;
    g1 = (rr1 < cnt) ? (off + rr1) : off;
    g2 = (rr2 < cnt) ? (off + rr2) : off;
    g3 = (rr3 < cnt) ? (off + rr3) : off;
  }
  const unsigned short* aP0 = A + (size_t)g0 * K + aCol;
  const unsigned short* aP1 = A + (size_t)g1 * K + aCol;
  const unsigned short* aP2 = A + (size_t)g2 * K + aCol;
  const unsigned short* aP3 = A + (size_t)g3 * K + aCol;
  const float* wP0 = W + (size_t)(n0 + wRow) * K + wCol;
  const float* wP1 = wP0 + (size_t)16 * K;
  const float* wP2 = wP0 + (size_t)32 * K;
  const float* wP3 = wP0 + (size_t)48 * K;
  const int aL = aRow * LDSS + aCol;
  const int wL = wRow * LDSS + wCol;

  v8f acc0 = {}, acc1 = {}, acc2 = {}, acc3 = {};
  uint4  ar0, ar1, ar2, ar3;
  float4 wr0, wr1, wr2, wr3;

  LOAD_TILE(0);
  int buf = 0;
  for (int k0 = 0; k0 < K; k0 += BK) {
    __syncthreads();
    STAGE_TILE(buf);
    if (k0 + BK < K) {
      LOAD_TILE(k0 + BK);
      if (k0 + 2 * BK < K)
        __builtin_prefetch(wP0 + (size_t)k0 + 2 * BK, 0, 1);
    }
    __syncthreads();
    compute_tile(&As[buf][0], &Bs[buf][0], wave, l16, hi, acc0, acc1, acc2, acc3);
    buf ^= 1;
  }
#pragma unroll
  for (int t = 0; t < 4; t++) {
    v8f av = (t == 0) ? acc0 : (t == 1) ? acc1 : (t == 2) ? acc2 : acc3;
#pragma unroll
    for (int r = 0; r < 8; r++) {
      int mrow = m0 + wave * 16 + r + hi * 8;
      if (mrow < cnt) {
        size_t slot = (size_t)(off + mrow);
        float val = av[r];
        if (WEIGHTED) val *= slot_wt[slot];
        C[slot * N + n0 + t * 16 + l16] = val;
      }
    }
  }
}

#undef LOAD_TILE
#undef STAGE_TILE

// ---------------- RMSNorm (row of kH) -> bf16 ----------------
__global__ __launch_bounds__(256) void rmsnorm_kernel(
    const float* __restrict__ x, const float* __restrict__ w,
    unsigned short* __restrict__ out) {
  const int t = blockIdx.x;
  const float* row = x + (size_t)t * kH;
  float v[8], ss = 0.f;
#pragma unroll
  for (int i = 0; i < 8; i++) { v[i] = row[threadIdx.x + i * 256]; ss += v[i] * v[i]; }
  for (int msk = 16; msk; msk >>= 1) ss += __shfl_xor(ss, msk, 32);
  __shared__ float red[8];
  int wv = threadIdx.x >> 5, ln = threadIdx.x & 31;
  if (ln == 0) red[wv] = ss;
  __syncthreads();
  float tot = 0.f;
#pragma unroll
  for (int i = 0; i < 8; i++) tot += red[i];
  float r = rsqrtf(tot / (float)kH + kEPS);
#pragma unroll
  for (int i = 0; i < 8; i++) {
    int c = threadIdx.x + i * 256;
    out[(size_t)t * kH + c] = f2bf(v[i] * r * w[c]);
  }
}

// ---------------- per-head QK RMSNorm + NeoX RoPE, V transpose ----------------
__global__ __launch_bounds__(128) void qkrope_kernel(
    const float* __restrict__ qkv, const int* __restrict__ pos,
    const float* __restrict__ qnw, const float* __restrict__ knw,
    unsigned short* __restrict__ qout,   // [NH][T][D]
    unsigned short* __restrict__ kout,   // [KVH][T][D]
    unsigned short* __restrict__ vtout)  // [KVH][D][T]
{
  const int t = blockIdx.x, hh = blockIdx.y, d = threadIdx.x;
  if (hh >= kNH + kKVH) {  // V: just transpose+convert
    int vh = hh - (kNH + kKVH);
    float val = qkv[(size_t)t * kQKVW + (kNH + kKVH) * kD + vh * kD + d];
    vtout[(size_t)vh * kD * kT + (size_t)d * kT + t] = f2bf(val);
    return;
  }
  const bool isq = hh < kNH;
  const int base = isq ? hh * kD : kNH * kD + (hh - kNH) * kD;
  float x = qkv[(size_t)t * kQKVW + base + d];
  float ss = x * x;
  for (int msk = 16; msk; msk >>= 1) ss += __shfl_xor(ss, msk, 32);
  __shared__ float red[4];
  __shared__ float xs[kD];
  int wv = threadIdx.x >> 5, ln = threadIdx.x & 31;
  if (ln == 0) red[wv] = ss;
  __syncthreads();
  float r = rsqrtf((red[0] + red[1] + red[2] + red[3]) / (float)kD + kEPS);
  float xn = x * r * (isq ? qnw[d] : knw[d]);
  xs[d] = xn;
  __syncthreads();
  int fi = d & 63;
  float inv_freq = __expf(-kLogTheta * (2.0f * (float)fi) / (float)kD);
  float ang = (float)pos[t] * inv_freq;
  float c = cosf(ang), s = sinf(ang);
  float rot = (d < 64) ? -xs[d + 64] : xs[d - 64];
  float o = c * xn + s * rot;
  if (isq) qout[(size_t)hh * kT * kD + (size_t)t * kD + d] = f2bf(o);
  else     kout[(size_t)(hh - kNH) * kT * kD + (size_t)t * kD + d] = f2bf(o);
}

// ---------------- flash attention (causal, GQA), bf16 WMMA ----------------
__global__ __launch_bounds__(256)
__attribute__((amdgpu_waves_per_eu(1)))
void attn_kernel(
    const unsigned short* __restrict__ q, const unsigned short* __restrict__ k,
    const unsigned short* __restrict__ vt, unsigned short* __restrict__ attn) {
  const int h = blockIdx.x, kvh = h >> 2;
  const int wave = threadIdx.x >> 5, lane = threadIdx.x & 31;
  const int hi = lane >> 4, l16 = lane & 15;
  const int q0 = blockIdx.y * 128 + wave * 16;
  __shared__ unsigned short P[8][16][32];

  const unsigned short* qbase = q + (size_t)h * kT * kD;
  const unsigned short* kbase = k + (size_t)kvh * kT * kD;
  const unsigned short* vbase = vt + (size_t)kvh * kD * kT;

  Frag qa[4];
#pragma unroll
  for (int kc = 0; kc < 4; kc++)
#pragma unroll
    for (int j = 0; j < 8; j++)
      qa[kc].u[j] = *(const unsigned int*)(qbase + (size_t)(q0 + l16) * kD + kc * 32 + afrag_k(j, hi));

  v8f o[8] = {};
  float mrow[8], lrow[8];
#pragma unroll
  for (int r = 0; r < 8; r++) { mrow[r] = -3.0e38f; lrow[r] = 0.f; }
  const float scale = 0.08838834764831845f;  // 1/sqrt(128)

  for (int kk = 0; kk < q0 + 16; kk += 32) {
    // batch-load all K fragments for this 32-key step (one wait covers all)
    Frag b0[4], b1[4];
#pragma unroll
    for (int kc = 0; kc < 4; kc++)
#pragma unroll
      for (int j = 0; j < 8; j++) {
        int kd = kc * 32 + bfrag_k(j, hi);
        b0[kc].u[j] = *(const unsigned int*)(kbase + (size_t)(kk + l16) * kD + kd);
        b1[kc].u[j] = *(const unsigned int*)(kbase + (size_t)(kk + 16 + l16) * kD + kd);
      }
    v8f s0 = {}, s1 = {};
#pragma unroll
    for (int kc = 0; kc < 4; kc++) {
      s0 = wmma_bf16(qa[kc], b0[kc], s0);
      s1 = wmma_bf16(qa[kc], b1[kc], s1);
    }
    // issue V fragments now: in flight during the softmax VALU work below
    Frag vb[8];
#pragma unroll
    for (int t = 0; t < 8; t++)
#pragma unroll
      for (int j = 0; j < 8; j++)
        vb[t].u[j] = *(const unsigned int*)(vbase + (size_t)(t * 16 + l16) * kT + kk + bfrag_k(j, hi));
#pragma unroll
    for (int r = 0; r < 8; r++) {
      int qi = q0 + r + hi * 8;
      float v0 = (kk + l16 <= qi) ? s0[r] * scale : -3.0e38f;
      float v1 = (kk + 16 + l16 <= qi) ? s1[r] * scale : -3.0e38f;
      float rm = fmaxf(v0, v1);
      for (int msk = 8; msk; msk >>= 1) rm = fmaxf(rm, __shfl_xor(rm, msk, 16));
      float mnew = fmaxf(mrow[r], rm);
      float p0 = __expf(v0 - mnew), p1 = __expf(v1 - mnew);
      float rs = p0 + p1;
      for (int msk = 8; msk; msk >>= 1) rs += __shfl_xor(rs, msk, 16);
      float corr = __expf(mrow[r] - mnew);
      lrow[r] = lrow[r] * corr + rs;
      mrow[r] = mnew;
#pragma unroll
      for (int t = 0; t < 8; t++) o[t][r] = o[t][r] * corr;
      P[wave][r + hi * 8][l16] = f2bf(p0);
      P[wave][r + hi * 8][16 + l16] = f2bf(p1);
    }
    // same-wave LDS store->load of P (DS ops are in-order; wait them out)
    asm volatile("s_wait_dscnt 0" ::: "memory");
    Frag pa;
#pragma unroll
    for (int j = 0; j < 8; j++)
      pa.u[j] = *(const unsigned int*)&P[wave][l16][afrag_k(j, hi)];
#pragma unroll
    for (int t = 0; t < 8; t++)
      o[t] = wmma_bf16(pa, vb[t], o[t]);
  }
#pragma unroll
  for (int r = 0; r < 8; r++) {
    int qi = q0 + r + hi * 8;
    float inv = 1.0f / lrow[r];
#pragma unroll
    for (int t = 0; t < 8; t++)
      attn[(size_t)qi * kH + h * kD + t * 16 + l16] = f2bf(o[t][r] * inv);
  }
}

// ---------------- router: softmax -> top-4 -> renorm ----------------
__global__ void router_kernel(const unsigned short* __restrict__ h2,
                              const float* __restrict__ rw,
                              int* __restrict__ topk_idx, float* __restrict__ topk_wt) {
  const int t = blockIdx.x, e = threadIdx.x;  // 32 threads = 32 experts
  const unsigned short* hrow = h2 + (size_t)t * kH;
  const float* wrow = rw + (size_t)e * kH;
  float acc = 0.f;
  for (int i = 0; i < kH; i += 4)
    acc += bf2f(hrow[i]) * wrow[i] + bf2f(hrow[i + 1]) * wrow[i + 1] +
           bf2f(hrow[i + 2]) * wrow[i + 2] + bf2f(hrow[i + 3]) * wrow[i + 3];
  float mx = acc;
  for (int msk = 16; msk; msk >>= 1) mx = fmaxf(mx, __shfl_xor(mx, msk, 32));
  float p = __expf(acc - mx);
  float sum = p;
  for (int msk = 16; msk; msk >>= 1) sum += __shfl_xor(sum, msk, 32);
  float prob = p / sum, cur = prob;
  float selw[4]; int seli[4];
#pragma unroll
  for (int kk = 0; kk < 4; kk++) {
    float v = cur; int i = e;
    for (int msk = 16; msk; msk >>= 1) {
      float ov = __shfl_xor(v, msk, 32);
      int   oi = __shfl_xor(i, msk, 32);
      if (ov > v || (ov == v && oi < i)) { v = ov; i = oi; }
    }
    selw[kk] = v; seli[kk] = i;
    if (e == i) cur = -1.0f;
  }
  float s4 = selw[0] + selw[1] + selw[2] + selw[3] + 1e-20f;
  if (e < 4) { topk_idx[t * 4 + e] = seli[e]; topk_wt[t * 4 + e] = selw[e] / s4; }
}

// ---------------- token->expert-slot scatter ----------------
__global__ void count_kernel(const int* __restrict__ idx, int* __restrict__ counts) {
  int i = blockIdx.x * blockDim.x + threadIdx.x;
  if (i < kSlots) atomicAdd(&counts[idx[i]], 1);
}
__global__ void prefix_kernel(const int* __restrict__ counts, int* __restrict__ offsets) {
  if (threadIdx.x == 0) {
    int a = 0;
    for (int e = 0; e < kE; e++) { offsets[e] = a; a += counts[e]; }
  }
}
__global__ void place_kernel(const int* __restrict__ idx, const float* __restrict__ wt,
                             const int* __restrict__ offsets, int* __restrict__ cursor,
                             int* __restrict__ perm, float* __restrict__ slot_wt,
                             int* __restrict__ slot_of) {
  int i = blockIdx.x * blockDim.x + threadIdx.x;
  if (i >= kSlots) return;
  int e = idx[i];
  int r = atomicAdd(&cursor[e], 1);
  int slot = offsets[e] + r;
  perm[slot] = i >> 2;
  slot_wt[slot] = wt[i];
  slot_of[i] = slot;
}

// ---------------- elementwise ----------------
__global__ void add_kernel(const float* __restrict__ a, const float* __restrict__ b,
                           float* __restrict__ o, int n) {
  int i = blockIdx.x * blockDim.x + threadIdx.x;
  if (i < n) o[i] = a[i] + b[i];
}
__global__ void act_kernel(const float* __restrict__ gate, const float* __restrict__ up,
                           unsigned short* __restrict__ act, int n) {
  int i = blockIdx.x * blockDim.x + threadIdx.x;
  if (i < n) {
    float g = gate[i];
    act[i] = f2bf((g / (1.0f + __expf(-g))) * up[i]);
  }
}
// deterministic combine: out[t] = hidden[t] + sum_k cslots[slot_of[t*4+k]]
__global__ __launch_bounds__(256) void final_kernel(
    const float* __restrict__ hidden, const float* __restrict__ cslots,
    const int* __restrict__ slot_of, float* __restrict__ out) {
  int t = blockIdx.x;
  int c = blockIdx.y * 256 + threadIdx.x;
  float acc = hidden[(size_t)t * kH + c];
#pragma unroll
  for (int kk = 0; kk < 4; kk++)
    acc += cslots[(size_t)slot_of[t * 4 + kk] * kH + c];
  out[(size_t)t * kH + c] = acc;
}

// ---------------- host ----------------
extern "C" void kernel_launch(void* const* d_in, const int* in_sizes, int n_in,
                              void* d_out, int out_size, void* d_ws, size_t ws_size,
                              hipStream_t stream) {
  (void)in_sizes; (void)n_in; (void)out_size; (void)ws_size;
  const float* x        = (const float*)d_in[0];
  const int*   pos      = (const int*)d_in[1];
  const float* ln1_w    = (const float*)d_in[2];
  const float* w_qkv    = (const float*)d_in[3];
  const float* qnw      = (const float*)d_in[4];
  const float* knw      = (const float*)d_in[5];
  const float* w_o      = (const float*)d_in[6];
  const float* ln2_w    = (const float*)d_in[7];
  const float* router_w = (const float*)d_in[8];
  const float* w_gate   = (const float*)d_in[9];
  const float* w_up     = (const float*)d_in[10];
  const float* w_down   = (const float*)d_in[11];
  float* out = (float*)d_out;

  char* p = (char*)d_ws;
  auto alloc = [&](size_t bytes) -> char* {
    char* r = p; p += (bytes + 255) & ~(size_t)255; return r;
  };
  unsigned short* h1   = (unsigned short*)alloc((size_t)kT * kH * 2);
  float*          qkv  = (float*)alloc((size_t)kT * kQKVW * 4);
  unsigned short* qb   = (unsigned short*)alloc((size_t)kNH * kT * kD * 2);
  unsigned short* kb   = (unsigned short*)alloc((size_t)kKVH * kT * kD * 2);
  unsigned short* vtb  = (unsigned short*)alloc((size_t)kKVH * kD * kT * 2);
  unsigned short* attn = (unsigned short*)alloc((size_t)kT * kH * 2);
  float*          aout = (float*)alloc((size_t)kT * kH * 4);
  float*          hid  = (float*)alloc((size_t)kT * kH * 4);
  unsigned short* h2   = (unsigned short*)alloc((size_t)kT * kH * 2);
  int*            tki  = (int*)alloc((size_t)kSlots * 4);
  float*          tkw  = (float*)alloc((size_t)kSlots * 4);
  int*            ctl  = (int*)alloc((size_t)3 * kE * 4);  // counts|offsets|cursor
  int*            perm = (int*)alloc((size_t)kSlots * 4);
  float*          swt  = (float*)alloc((size_t)kSlots * 4);
  int*            sof  = (int*)alloc((size_t)kSlots * 4);
  float*          gbuf = (float*)alloc((size_t)kSlots * kI * 4);
  float*          ubuf = (float*)alloc((size_t)kSlots * kI * 4);
  unsigned short* abuf = (unsigned short*)alloc((size_t)kSlots * kI * 2);
  float*          csl  = (float*)alloc((size_t)kSlots * kH * 4);
  int* counts = ctl, *offsets = ctl + kE, *cursor = ctl + 2 * kE;

  // ---- attention block ----
  rmsnorm_kernel<<<kT, 256, 0, stream>>>(x, ln1_w, h1);
  gemm_bf16_kernel<<<dim3(kQKVW / BN, kT / BM), 256, 0, stream>>>(h1, w_qkv, qkv, kT, kQKVW, kH);
  qkrope_kernel<<<dim3(kT, kNH + 2 * kKVH), 128, 0, stream>>>(qkv, pos, qnw, knw, qb, kb, vtb);
  attn_kernel<<<dim3(kNH, kT / 128), 256, 0, stream>>>(qb, kb, vtb, attn);
  gemm_bf16_kernel<<<dim3(kH / BN, kT / BM), 256, 0, stream>>>(attn, w_o, aout, kT, kH, kH);
  add_kernel<<<(kT * kH) / 256, 256, 0, stream>>>(x, aout, hid, kT * kH);

  // ---- MoE block ----
  rmsnorm_kernel<<<kT, 256, 0, stream>>>(hid, ln2_w, h2);
  router_kernel<<<kT, 32, 0, stream>>>(h2, router_w, tki, tkw);
  hipMemsetAsync(ctl, 0, 3 * kE * sizeof(int), stream);
  count_kernel<<<(kSlots + 255) / 256, 256, 0, stream>>>(tki, counts);
  prefix_kernel<<<1, 32, 0, stream>>>(counts, offsets);
  place_kernel<<<(kSlots + 255) / 256, 256, 0, stream>>>(tki, tkw, offsets, cursor, perm, swt, sof);

  moe_gemm_kernel<true, false><<<dim3(kI / BN, kT / BM, kE), 256, 0, stream>>>(
      h2, w_gate, gbuf, kI, kH, (size_t)kI * kH, counts, offsets, perm, swt);
  moe_gemm_kernel<true, false><<<dim3(kI / BN, kT / BM, kE), 256, 0, stream>>>(
      h2, w_up, ubuf, kI, kH, (size_t)kI * kH, counts, offsets, perm, swt);
  act_kernel<<<(kSlots * kI) / 256, 256, 0, stream>>>(gbuf, ubuf, abuf, kSlots * kI);
  moe_gemm_kernel<false, true><<<dim3(kH / BN, kT / BM, kE), 256, 0, stream>>>(
      abuf, w_down, csl, kH, kI, (size_t)kH * kI, counts, offsets, perm, swt);
  final_kernel<<<dim3(kT, kH / 256), 256, 0, stream>>>(hid, csl, sof, out);
}
